// MotherNet_89498528514766
// MI455X (gfx1250) — matrix operand order repaired and checked by reading, not compiled, not run on hardware
//
#include <hip/hip_runtime.h>

typedef _Float16 v16h __attribute__((ext_vector_type(16)));
typedef float    v8f  __attribute__((ext_vector_type(8)));

#define T_TOT   8192
#define BATCH   8
#define F_IN    100
#define KP1     128          // F zero-padded to multiple of 32
#define KG1     (KP1 / 32)   // 4 k-groups, layer 1
#define H_DIM   512
#define KG2     (H_DIM / 32) // 16 k-groups, layer 2
#define TM      64           // T rows per workgroup
#define XSTR    136          // f16 stride (272 B = 4 words mod 64 banks)
#define H1STR   520          // f16 stride (1040 B = 4 words mod 64 banks)
#define NWAVES  8
#define WAVE_N  64
// packed-weight geometry: per k-group: 32 col-tiles * 32 lanes * 16 halves
#define PK_KG_STRIDE 16384   // halves per k-group slab
#define W1P_PER_B (KG1 * PK_KG_STRIDE)  //  65,536 halves
#define W2P_PER_B (KG2 * PK_KG_STRIDE)  // 262,144 halves
#define W2P_OFFSET_BYTES (BATCH * W1P_PER_B * 2)  // 1 MiB

union Frag16 { v16h h; uint4 q[2]; };

__device__ __forceinline__ float sanitize(float v) {
    const float M = 3.4028234663852886e38f;   // jnp.nan_to_num
    if (v != v) return 0.f;
    if (v >  M) return  M;
    if (v < -M) return -M;
    return v;
}

// Repack [B, K, 512] f32 weights into per-lane B-fragment order, f16:
// idx = (((b*KG + kg)*32 + ntile)*32 + lane)*16 + e
//   col = ntile*16 + (lane&15) ; k = kg*32 + (lane>>4)*8 + e + (e&8)  (0 if k >= K)
__global__ void pack_weights(const float* __restrict__ src, _Float16* __restrict__ dst,
                             int K, int total) {
    for (int i = blockIdx.x * blockDim.x + threadIdx.x; i < total;
         i += gridDim.x * blockDim.x) {
        int e    = i & 15;
        int lane = (i >> 4) & 31;
        int nt   = (i >> 9) & 31;
        int kgb  = i >> 14;            // b*KG + kg (PK_KG_STRIDE == 2^14)
        int col  = nt * 16 + (lane & 15);
        int k    = kgb * 32 + (lane >> 4) * 8 + e + (e & 8);   // b*KGpad*32 + k_local
        // split b / k_local using padded K per matrix:
        int Kpad = (K == F_IN) ? KP1 : H_DIM;
        int b  = k / Kpad;
        int kl = k - b * Kpad;
        float v = (kl < K) ? src[((size_t)b * K + kl) * H_DIM + col] : 0.f;
        dst[i] = (_Float16)v;
    }
}

__global__ __launch_bounds__(256, 1)
void mothernet_fused(const float* __restrict__ x,
                     const _Float16* __restrict__ w1p, const float* __restrict__ b1,
                     const _Float16* __restrict__ w2p, const float* __restrict__ b2,
                     const float* __restrict__ w3, const float* __restrict__ b3,
                     float* __restrict__ out)
{
    __shared__ __align__(16) _Float16 xs[TM * XSTR];    // 17,408 B
    __shared__ __align__(16) _Float16 h1s[TM * H1STR];  // 66,560 B
    __shared__ float partial[NWAVES * TM];              //  2,048 B

    const int bid  = blockIdx.x;
    const int b    = bid & (BATCH - 1);
    const int t0   = (bid >> 3) * TM;
    const int tid  = threadIdx.x;
    const int wave = tid >> 5;
    const int lane = tid & 31;
    const int l15  = lane & 15;
    const int hi   = lane >> 4;

    // ---------------- Phase 0: stage sanitized x tile as f16 ----------------
    {
        const float* xg = x + (size_t)t0 * (BATCH * F_IN) + (size_t)b * F_IN;
        for (int i = tid; i < TM * KP1; i += 256) {
            int r = i >> 7;           // / KP1
            int f = i & (KP1 - 1);
            float v = (f < F_IN) ? sanitize(xg[(size_t)r * (BATCH * F_IN) + f]) : 0.f;
            xs[r * XSTR + f] = (_Float16)v;
        }
    }
    __syncthreads();

    const int n_base = wave * WAVE_N;

    // ---------------- Phase 1: h1 = relu(x @ w1 + b1) ----------------
    {
        v8f acc[4][4];
        #pragma unroll
        for (int m = 0; m < 4; ++m)
            #pragma unroll
            for (int nt = 0; nt < 4; ++nt)
                acc[m][nt] = (v8f){0.f,0.f,0.f,0.f,0.f,0.f,0.f,0.f};

        const _Float16* w1b = w1p + (size_t)b * W1P_PER_B;

        #pragma unroll
        for (int kg = 0; kg < KG1; ++kg) {
            const int k0 = kg * 32;
            Frag16 afr[4], bfr[4];
            #pragma unroll
            for (int m = 0; m < 4; ++m) {
                const _Float16* p = &xs[(m * 16 + l15) * XSTR + k0 + hi * 8];
                afr[m].q[0] = *(const uint4*)(p);
                afr[m].q[1] = *(const uint4*)(p + 16);
            }
            #pragma unroll
            for (int nt = 0; nt < 4; ++nt) {
                const _Float16* p = w1b + ((size_t)kg * PK_KG_STRIDE)
                                  + (((wave * 4 + nt) * 32 + lane) * 16);
                bfr[nt].q[0] = *(const uint4*)(p);
                bfr[nt].q[1] = *(const uint4*)(p + 8);
            }
            #pragma unroll
            for (int m = 0; m < 4; ++m)
                #pragma unroll
                for (int nt = 0; nt < 4; ++nt)
                    acc[m][nt] = __builtin_amdgcn_wmma_f32_16x16x32_f16(
                        false, afr[m].h, false, bfr[nt].h,
                        (short)0, acc[m][nt], false, false);
        }

        const float* b1g = b1 + b * H_DIM;
        #pragma unroll
        for (int nt = 0; nt < 4; ++nt) {
            int col = n_base + nt * 16 + l15;
            float bias = b1g[col];
            #pragma unroll
            for (int m = 0; m < 4; ++m) {
                #pragma unroll
                for (int r = 0; r < 8; ++r) {
                    int row = m * 16 + hi * 8 + r;      // C/D layout
                    float v = acc[m][nt][r] + bias;
                    h1s[row * H1STR + col] = (_Float16)(v > 0.f ? v : 0.f);
                }
            }
        }
    }
    __syncthreads();

    // -------- Phase 2: h2 = relu(h1 @ w2 + b2), fused with layer 3 --------
    {
        v8f acc[4][4];
        #pragma unroll
        for (int m = 0; m < 4; ++m)
            #pragma unroll
            for (int nt = 0; nt < 4; ++nt)
                acc[m][nt] = (v8f){0.f,0.f,0.f,0.f,0.f,0.f,0.f,0.f};

        const _Float16* w2b = w2p + (size_t)b * W2P_PER_B;

        for (int kg = 0; kg < KG2; ++kg) {
            const int k0 = kg * 32;
            Frag16 afr[4], bfr[4];
            #pragma unroll
            for (int m = 0; m < 4; ++m) {
                const _Float16* p = &h1s[(m * 16 + l15) * H1STR + k0 + hi * 8];
                afr[m].q[0] = *(const uint4*)(p);
                afr[m].q[1] = *(const uint4*)(p + 16);
            }
            #pragma unroll
            for (int nt = 0; nt < 4; ++nt) {
                const _Float16* p = w2b + ((size_t)kg * PK_KG_STRIDE)
                                  + (((wave * 4 + nt) * 32 + lane) * 16);
                bfr[nt].q[0] = *(const uint4*)(p);
                bfr[nt].q[1] = *(const uint4*)(p + 8);
            }
            #pragma unroll
            for (int m = 0; m < 4; ++m)
                #pragma unroll
                for (int nt = 0; nt < 4; ++nt)
                    acc[m][nt] = __builtin_amdgcn_wmma_f32_16x16x32_f16(
                        false, afr[m].h, false, bfr[nt].h,
                        (short)0, acc[m][nt], false, false);
        }

        // Epilogue: +b2, relu, dot with w3 (layer 3), in-wave reduce
        const float* b2g = b2 + b * H_DIM;
        const float* w3g = w3 + b * H_DIM;
        float rowpart[4][8];
        #pragma unroll
        for (int m = 0; m < 4; ++m)
            #pragma unroll
            for (int r = 0; r < 8; ++r)
                rowpart[m][r] = 0.f;

        #pragma unroll
        for (int nt = 0; nt < 4; ++nt) {
            int col = n_base + nt * 16 + l15;
            float bias = b2g[col];
            float w3v  = w3g[col];
            #pragma unroll
            for (int m = 0; m < 4; ++m) {
                #pragma unroll
                for (int r = 0; r < 8; ++r) {
                    float v = acc[m][nt][r] + bias;
                    v = v > 0.f ? v : 0.f;
                    rowpart[m][r] += v * w3v;
                }
            }
        }

        #pragma unroll
        for (int m = 0; m < 4; ++m) {
            #pragma unroll
            for (int r = 0; r < 8; ++r) {
                float v = rowpart[m][r];
                v += __shfl_xor(v, 1, 32);
                v += __shfl_xor(v, 2, 32);
                v += __shfl_xor(v, 4, 32);
                v += __shfl_xor(v, 8, 32);
                if (l15 == 0)
                    partial[wave * TM + m * 16 + hi * 8 + r] = v;
            }
        }
    }
    __syncthreads();

    // ---------------- Phase 3: cross-wave reduce, write out ----------------
    if (tid < TM) {
        float s = b3[b];
        #pragma unroll
        for (int w = 0; w < NWAVES; ++w)
            s += partial[w * TM + tid];
        out[(size_t)(t0 + tid) * BATCH + b] = s;
    }
}

extern "C" void kernel_launch(void* const* d_in, const int* in_sizes, int n_in,
                              void* d_out, int out_size, void* d_ws, size_t ws_size,
                              hipStream_t stream) {
    (void)in_sizes; (void)n_in; (void)ws_size; (void)out_size;
    const float* x  = (const float*)d_in[0];
    const float* w1 = (const float*)d_in[1];
    const float* b1 = (const float*)d_in[2];
    const float* w2 = (const float*)d_in[3];
    const float* b2 = (const float*)d_in[4];
    const float* w3 = (const float*)d_in[5];
    const float* b3 = (const float*)d_in[6];
    float* out = (float*)d_out;

    _Float16* w1p = (_Float16*)d_ws;
    _Float16* w2p = (_Float16*)((char*)d_ws + W2P_OFFSET_BYTES);

    // Pre-pack weights into f16 B-fragment layout (runs every call; ~1 us)
    const int total1 = BATCH * W1P_PER_B;   //   524,288 halves
    const int total2 = BATCH * W2P_PER_B;   // 2,097,152 halves
    pack_weights<<<128, 256, 0, stream>>>(w1, w1p, F_IN,  total1);
    pack_weights<<<256, 256, 0, stream>>>(w2, w2p, H_DIM, total2);

    dim3 grid((T_TOT / TM) * BATCH);   // 1024 workgroups
    dim3 block(256);                   // 8 waves (wave32)
    mothernet_fused<<<grid, block, 0, stream>>>(x, w1p, b1, w2p, b2, w3, b3, out);
}
